// SandboxUnit_13821204758961
// MI455X (gfx1250) — matrix operand order repaired and checked by reading, not compiled, not run on hardware
//
#include <hip/hip_runtime.h>
#include <math.h>

// ---------------- types & constants ----------------
typedef __attribute__((ext_vector_type(16))) __bf16 v16bf;
typedef __attribute__((ext_vector_type(8)))  __bf16 v8bf;
typedef __attribute__((ext_vector_type(8)))  float  v8f;

constexpr int Bc   = 2;
constexpr int Sc   = 2048;
constexpr int Dc   = 2048;
constexpr int Hc   = 16;
constexpr int HDc  = 128;
constexpr int HIDc = 8192;
constexpr int Mc   = Bc * Sc;          // 4096 rows of activations

#define WMMA_BF16(a, b, c) \
  __builtin_amdgcn_wmma_f32_16x16x32_bf16(false, (a), false, (b), (short)0, (c), false, false)

// ---------------- small helpers ----------------
__device__ __forceinline__ __bf16 f2bf(float f) {
  union { float f; unsigned u; } a; a.f = f;
  unsigned r = a.u + 0x7FFFu + ((a.u >> 16) & 1u);   // round-to-nearest-even
  union { unsigned short s; __bf16 b; } o; o.s = (unsigned short)(r >> 16);
  return o.b;
}

// A-matrix fragment (16x32 bf16): lane m = lane&15, g = lane>>4.
// element e -> K = (e<8) ? e+8g : e+8+8g   (interleaved halves, per ISA table)
__device__ __forceinline__ v16bf load_a_frag(const __bf16* __restrict__ p, int g) {
  v8bf lo = *(const v8bf*)(p + 8 * g);
  v8bf hi = *(const v8bf*)(p + 16 + 8 * g);
  v16bf f;
#pragma unroll
  for (int e = 0; e < 8; ++e) { f[e] = lo[e]; f[e + 8] = hi[e]; }
  return f;
}

// B-matrix fragment (32x16 bf16): lane n = lane&15, g = lane>>4.
// element e -> K = (e<8) ? e+8g : e+8+8g   (interleaved halves, matches A)
__device__ __forceinline__ v16bf load_b_frag(const __bf16* __restrict__ p, int g) {
  v8bf lo = *(const v8bf*)(p + 8 * g);
  v8bf hi = *(const v8bf*)(p + 16 + 8 * g);
  v16bf f;
#pragma unroll
  for (int e = 0; e < 8; ++e) { f[e] = lo[e]; f[e + 8] = hi[e]; }
  return f;
}

// ---------------- weight convert + transpose: f32[K][N] -> bf16[N][K] ----------------
__global__ void k_wconv(const float* __restrict__ in, __bf16* __restrict__ out,
                        int Kdim, int Ndim) {
  long long idx = (long long)blockIdx.x * blockDim.x + threadIdx.x;
  long long total = (long long)Kdim * Ndim;
  if (idx >= total) return;
  int k = (int)(idx / Ndim);
  int n = (int)(idx % Ndim);
  out[(size_t)n * Kdim + k] = f2bf(in[idx]);
}

// ---------------- centered RMSNorm: fp32 row -> bf16 row ----------------
__global__ __launch_bounds__(256) void k_rmsnorm(const float* __restrict__ x,
                                                 const float* __restrict__ w,
                                                 __bf16* __restrict__ out, int D) {
  __shared__ float sred[8];
  const int row = blockIdx.x;
  const size_t base = (size_t)row * D;
  const int tid = threadIdx.x, lane = tid & 31, wv = tid >> 5;
  float ss = 0.f;
  for (int i = tid; i < D; i += 256) { float v = x[base + i]; ss += v * v; }
#pragma unroll
  for (int off = 16; off >= 1; off >>= 1) ss += __shfl_xor(ss, off, 32);
  if (lane == 0) sred[wv] = ss;
  __syncthreads();
  if (tid == 0) {
    float t = 0.f;
#pragma unroll
    for (int j = 0; j < 8; ++j) t += sred[j];
    sred[0] = rsqrtf(t / (float)D + 1e-6f);
  }
  __syncthreads();
  const float rs = sred[0];
  for (int i = tid; i < D; i += 256)
    out[base + i] = f2bf((1.0f + w[i]) * (x[base + i] * rs));
}

// ---------------- generic bf16 WMMA GEMM: C = A[MxK] * Bt[NxK]^T ----------------
// MODE: 0=Q (scale, per-head [b,h,s,d])  1=K (same)  2=V (transposed [b,h,d,s])
//       3=f32 out + residual             4=SwiGLU dual-B -> bf16
// MT: number of 16-row M tiles per wave (B fragments reused MT times).
template <int MODE, int MT>
__global__ __launch_bounds__(128) void k_gemm(const __bf16* __restrict__ A,
                                              const __bf16* __restrict__ Bt,
                                              const __bf16* __restrict__ Bt2,
                                              const float* __restrict__ resid,
                                              float* __restrict__ outf,
                                              __bf16* __restrict__ outb,
                                              int M, int N, int K, float scale) {
  constexpr bool DUAL = (MODE == 4);
  const int lane = threadIdx.x & 31;
  const int wv   = threadIdx.x >> 5;
  const int g    = lane >> 4;
  const int r16  = lane & 15;
  const int m0   = blockIdx.x * (16 * MT);
  const int n0   = blockIdx.y * 256 + wv * 64;

  const __bf16* arow[MT];
#pragma unroll
  for (int mt = 0; mt < MT; ++mt)
    arow[mt] = A + (size_t)(m0 + mt * 16 + r16) * K;

  const __bf16* brow[4];
  const __bf16* b2row[4];
#pragma unroll
  for (int t = 0; t < 4; ++t) {
    brow[t] = Bt + (size_t)(n0 + t * 16 + r16) * K;
    if constexpr (DUAL) b2row[t] = Bt2 + (size_t)(n0 + t * 16 + r16) * K;
  }

  v8f acc[MT][4];
#pragma unroll
  for (int mt = 0; mt < MT; ++mt)
#pragma unroll
    for (int t = 0; t < 4; ++t) acc[mt][t] = (v8f){};
  v8f acc1[4];
  if constexpr (DUAL) {
#pragma unroll
    for (int t = 0; t < 4; ++t) acc1[t] = (v8f){};
  }

  for (int k0 = 0; k0 < K; k0 += 32) {
    // issue ALL fragment loads first so the backend can clause them
    v16bf af[MT];
#pragma unroll
    for (int mt = 0; mt < MT; ++mt) af[mt] = load_a_frag(arow[mt] + k0, g);
    v16bf bfr[4];
#pragma unroll
    for (int t = 0; t < 4; ++t) {
      __builtin_prefetch(brow[t] + k0 + 256, 0, 1);
      bfr[t] = load_b_frag(brow[t] + k0, g);
    }
    v16bf bfr2[4];
    if constexpr (DUAL) {
#pragma unroll
      for (int t = 0; t < 4; ++t) bfr2[t] = load_b_frag(b2row[t] + k0, g);
    }
    // then the WMMA chain
#pragma unroll
    for (int t = 0; t < 4; ++t) {
#pragma unroll
      for (int mt = 0; mt < MT; ++mt)
        acc[mt][t] = WMMA_BF16(af[mt], bfr[t], acc[mt][t]);
      if constexpr (DUAL)
        acc1[t] = WMMA_BF16(af[0], bfr2[t], acc1[t]);
    }
  }

  // epilogue; C layout: VGPR r, lane l -> M = r + 8*(l>>4), N = l&15
#pragma unroll
  for (int mt = 0; mt < MT; ++mt) {
#pragma unroll
    for (int t = 0; t < 4; ++t) {
#pragma unroll
      for (int r = 0; r < 8; ++r) {
        const int m = m0 + mt * 16 + r + 8 * g;
        const int n = n0 + t * 16 + r16;
        float v = acc[mt][t][r];
        if constexpr (MODE == 0 || MODE == 1) {      // Q/K: scale + [b,h,s,d]
          v *= scale;
          const int h = n >> 7, d = n & (HDc - 1);
          const int b = m >> 11, s = m & (Sc - 1);
          outb[(((size_t)b * Hc + h) * Sc + s) * HDc + d] = f2bf(v);
        } else if constexpr (MODE == 2) {            // V: [b,h,d,s] (transposed)
          const int h = n >> 7, d = n & (HDc - 1);
          const int b = m >> 11, s = m & (Sc - 1);
          outb[(((size_t)b * Hc + h) * HDc + d) * Sc + s] = f2bf(v);
        } else if constexpr (MODE == 3) {            // fp32 out + residual
          const size_t idx = (size_t)m * N + n;
          outf[idx] = v + resid[idx];
        } else {                                      // SwiGLU: silu(a)*b -> bf16
          const float a  = v;
          const float b1 = acc1[t][r];
          const float si = a / (1.0f + __expf(-a));
          outb[(size_t)m * N + n] = f2bf(si * b1);
        }
      }
    }
  }
}

// ---------------- attention: one wave per (b,h, 16-query tile) ----------------
// q,k: [b,h,s,d] bf16 (pre-scaled by d^-1/4); vt: [b,h,d,s] bf16; o: [b,s,h*hd] bf16
__global__ __launch_bounds__(32) void k_attn(const __bf16* __restrict__ qb,
                                             const __bf16* __restrict__ kb,
                                             const __bf16* __restrict__ vt,
                                             __bf16* __restrict__ ob) {
  extern __shared__ float smem[];          // [16][Sc] scores + [16] inv-sums
  float* sp   = smem;
  float* sinv = smem + 16 * Sc;

  const int blk  = blockIdx.x;
  const int qt   = blk % (Sc / 16);
  const int bh   = blk / (Sc / 16);
  const int lane = threadIdx.x & 31;
  const int g    = lane >> 4;
  const int r16  = lane & 15;
  const int qbase = qt * 16;
  const int nkeys = (qt + 1) * 16;

  const __bf16* qhead = qb + (size_t)bh * Sc * HDc;
  const __bf16* khead = kb + (size_t)bh * Sc * HDc;
  const __bf16* vhead = vt + (size_t)bh * HDc * Sc;

  // Q fragments over head-dim (4 chunks of 32)
  v16bf qf[4];
  {
    const __bf16* qrow = qhead + (size_t)(qbase + r16) * HDc;
#pragma unroll
    for (int c = 0; c < 4; ++c) qf[c] = load_a_frag(qrow + 32 * c, g);
  }

  // pass 1: scores -> LDS (causal mask on diagonal tile)
  for (int kt = 0; kt <= qt; ++kt) {
    v8f sc = {};
    const __bf16* krow = khead + (size_t)(kt * 16 + r16) * HDc;
    v16bf kf[4];
#pragma unroll
    for (int c = 0; c < 4; ++c) kf[c] = load_b_frag(krow + 32 * c, g);
#pragma unroll
    for (int c = 0; c < 4; ++c) sc = WMMA_BF16(qf[c], kf[c], sc);
    const int key = kt * 16 + r16;
#pragma unroll
    for (int r = 0; r < 8; ++r) {
      const int qr = r + 8 * g;
      float v = sc[r];
      if (kt == qt && key > qbase + qr) v = -3.0e38f;   // causal mask
      sp[qr * Sc + key] = v;
    }
  }
  __syncthreads();   // single-wave WG: lowers to S_NOP, DS ordering is in-order anyway

  // softmax per row (causal Z; reference's full-row Z + tril + renorm reduces to this)
  for (int rr = 0; rr < 16; ++rr) {
    float mx = -3.0e38f;
    for (int col = lane; col < nkeys; col += 32) mx = fmaxf(mx, sp[rr * Sc + col]);
#pragma unroll
    for (int off = 16; off >= 1; off >>= 1) mx = fmaxf(mx, __shfl_xor(mx, off, 32));
    float sum = 0.f;
    for (int col = lane; col < nkeys; col += 32) {
      float e = __expf(sp[rr * Sc + col] - mx);
      sp[rr * Sc + col] = e;
      sum += e;
    }
#pragma unroll
    for (int off = 16; off >= 1; off >>= 1) sum += __shfl_xor(sum, off, 32);
    if (lane == 0) sinv[rr] = 1.0f / sum;
  }
  __syncthreads();

  // pass 2: O = P @ V using V^T rows (d contiguous in keys)
  v8f oacc[8] = {};
  const float inv = sinv[r16];
  const int nsteps = (nkeys + 31) / 32;
  for (int ks = 0; ks < nsteps; ++ks) {
    const int kbase = ks * 32;
    v16bf pf;
#pragma unroll
    for (int e = 0; e < 16; ++e) {
      const int key = kbase + ((e < 8) ? (e + 8 * g) : (e + 8 + 8 * g));
      const float pv = (key < nkeys) ? sp[r16 * Sc + key] * inv : 0.f;
      pf[e] = f2bf(pv);
    }
#pragma unroll
    for (int n = 0; n < 8; ++n) {
      const __bf16* vrow = vhead + (size_t)(n * 16 + r16) * Sc + kbase;
      v16bf vf = load_b_frag(vrow, g);
      oacc[n] = WMMA_BF16(pf, vf, oacc[n]);
    }
  }

  // store O: [b, s, h*HD + d]
  const int b = bh / Hc, h = bh % Hc;
#pragma unroll
  for (int n = 0; n < 8; ++n) {
#pragma unroll
    for (int r = 0; r < 8; ++r) {
      const int qr = r + 8 * g;
      const int d  = n * 16 + r16;
      ob[((size_t)(b * Sc + qbase + qr)) * Dc + h * HDc + d] = f2bf(oacc[n][r]);
    }
  }
}

// ---------------- host launcher ----------------
extern "C" void kernel_launch(void* const* d_in, const int* in_sizes, int n_in,
                              void* d_out, int out_size, void* d_ws, size_t ws_size,
                              hipStream_t stream) {
  (void)in_sizes; (void)n_in; (void)out_size; (void)ws_size;
  const float* x      = (const float*)d_in[0];
  const float* ln_w   = (const float*)d_in[1];
  const float* ffln_w = (const float*)d_in[2];
  const float* wq     = (const float*)d_in[3];
  const float* wk     = (const float*)d_in[4];
  const float* wv     = (const float*)d_in[5];
  const float* wo     = (const float*)d_in[6];
  const float* wg     = (const float*)d_in[7];
  const float* w1     = (const float*)d_in[8];
  const float* w2     = (const float*)d_in[9];
  float* out = (float*)d_out;

  // workspace carve-up (256B aligned)
  char* p = (char*)d_ws;
  auto carve = [&](size_t bytes) -> void* {
    void* r = (void*)p;
    p += (bytes + 255) & ~(size_t)255;
    return r;
  };
  __bf16* wq_t = (__bf16*)carve((size_t)Dc * Dc * 2);
  __bf16* wk_t = (__bf16*)carve((size_t)Dc * Dc * 2);
  __bf16* wv_t = (__bf16*)carve((size_t)Dc * Dc * 2);
  __bf16* wo_t = (__bf16*)carve((size_t)Dc * Dc * 2);
  __bf16* wg_t = (__bf16*)carve((size_t)Dc * HIDc * 2);
  __bf16* w1_t = (__bf16*)carve((size_t)Dc * HIDc * 2);
  __bf16* w2_t = (__bf16*)carve((size_t)HIDc * Dc * 2);
  __bf16* hbuf = (__bf16*)carve((size_t)Mc * Dc * 2);
  __bf16* qbuf = (__bf16*)carve((size_t)Mc * Dc * 2);
  __bf16* kbuf = (__bf16*)carve((size_t)Mc * Dc * 2);
  __bf16* vtb  = (__bf16*)carve((size_t)Mc * Dc * 2);
  __bf16* obuf = (__bf16*)carve((size_t)Mc * Dc * 2);
  float*  x2   = (float*) carve((size_t)Mc * Dc * 4);
  __bf16* h2   = (__bf16*)carve((size_t)Mc * Dc * 2);
  __bf16* gbuf = (__bf16*)carve((size_t)Mc * HIDc * 2);

  // 1) weight convert + transpose
  auto conv = [&](const float* in, __bf16* o, int Kd, int Nd) {
    long long tot = (long long)Kd * Nd;
    int blocks = (int)((tot + 255) / 256);
    k_wconv<<<blocks, 256, 0, stream>>>(in, o, Kd, Nd);
  };
  conv(wq, wq_t, Dc, Dc);
  conv(wk, wk_t, Dc, Dc);
  conv(wv, wv_t, Dc, Dc);
  conv(wo, wo_t, Dc, Dc);
  conv(wg, wg_t, Dc, HIDc);
  conv(w1, w1_t, Dc, HIDc);
  conv(w2, w2_t, HIDc, Dc);

  // 2) pre-attention RMSNorm
  k_rmsnorm<<<Mc, 256, 0, stream>>>(x, ln_w, hbuf, Dc);

  const float qk_scale = 0.29730177875068026f;  // 128^(-1/4)

  // 3) QKV projections (MT=2: each wave does 32 M rows x 64 N)
  {
    dim3 grid(Mc / 32, Dc / 256), block(128);
    k_gemm<0, 2><<<grid, block, 0, stream>>>(hbuf, wq_t, nullptr, nullptr, nullptr,
                                             qbuf, Mc, Dc, Dc, qk_scale);
    k_gemm<1, 2><<<grid, block, 0, stream>>>(hbuf, wk_t, nullptr, nullptr, nullptr,
                                             kbuf, Mc, Dc, Dc, qk_scale);
    k_gemm<2, 2><<<grid, block, 0, stream>>>(hbuf, wv_t, nullptr, nullptr, nullptr,
                                             vtb, Mc, Dc, Dc, 1.0f);
  }

  // 4) attention (causal softmax, scores in LDS)
  {
    const size_t shmem = (size_t)(16 * Sc + 16) * sizeof(float);
    k_attn<<<Bc * Hc * (Sc / 16), 32, shmem, stream>>>(qbuf, kbuf, vtb, obuf);
  }

  // 5) output projection + residual -> x2 (fp32)
  {
    dim3 grid(Mc / 32, Dc / 256), block(128);
    k_gemm<3, 2><<<grid, block, 0, stream>>>(obuf, wo_t, nullptr, x, x2, nullptr,
                                             Mc, Dc, Dc, 1.0f);
  }

  // 6) FFN RMSNorm
  k_rmsnorm<<<Mc, 256, 0, stream>>>(x2, ffln_w, h2, Dc);

  // 7) fused SwiGLU gate+up GEMM (dual-B, MT=1)
  {
    dim3 grid(Mc / 16, HIDc / 256), block(128);
    k_gemm<4, 1><<<grid, block, 0, stream>>>(h2, wg_t, w1_t, nullptr, nullptr, gbuf,
                                             Mc, HIDc, Dc, 1.0f);
  }

  // 8) down projection + residual -> out (fp32)
  {
    dim3 grid(Mc / 32, Dc / 256), block(128);
    k_gemm<3, 2><<<grid, block, 0, stream>>>(gbuf, w2_t, nullptr, x2, out, nullptr,
                                             Mc, Dc, HIDc, 1.0f);
  }
}
